// GCN19_20693152432430
// MI455X (gfx1250) — compile-verified, matrix operand order checked by reading, not run on hardware
//
#include <hip/hip_runtime.h>
#include <hip/hip_bf16.h>

// ---------------- problem constants (match reference) ----------------
#define NN 100000      // nodes
#define EE 1600000     // edges
#define FD 128         // in features
#define HD 128         // hidden
#define GG 512         // graphs
#define BN_EPS 1e-5f

typedef __attribute__((ext_vector_type(16))) __bf16 v16bf;
typedef __attribute__((ext_vector_type(8)))  float  v8f;

// ---------------- small elementwise helpers ----------------
__global__ void k_fill_f32(float* __restrict__ p, float v, int n) {
    int i = blockIdx.x * blockDim.x + threadIdx.x;
    if (i < n) p[i] = v;
}

__global__ void k_deg(const int* __restrict__ dst, float* __restrict__ deg, int e) {
    int i = blockIdx.x * blockDim.x + threadIdx.x;
    if (i < e) atomicAdd(&deg[dst[i]], 1.0f);
}

__global__ void k_dinv(const float* __restrict__ deg, float* __restrict__ dinv, int n) {
    int i = blockIdx.x * blockDim.x + threadIdx.x;
    if (i < n) dinv[i] = rsqrtf(fmaxf(deg[i], 1.0f));
}

__global__ void k_norm(const int* __restrict__ src, const int* __restrict__ dst,
                       const float* __restrict__ dinv, float* __restrict__ norm, int e) {
    int i = blockIdx.x * blockDim.x + threadIdx.x;
    if (i < e) norm[i] = dinv[src[i]] * dinv[dst[i]];
}

__global__ void k_cvt_bf16(const float* __restrict__ x, __bf16* __restrict__ xb, int n) {
    int i = blockIdx.x * blockDim.x + threadIdx.x;
    if (i < n) xb[i] = (__bf16)x[i];
}

// Pack W [128 x 128 row-major, W[k][n]] into per-lane WMMA B-operand layout:
// Wp[((tileN*4 + kstep)*32 + lane)*16 + elem]  (each lane's 16 bf16 contiguous)
__global__ void k_pack_w(const float* __restrict__ W, __bf16* __restrict__ Wp) {
    int i = blockIdx.x * blockDim.x + threadIdx.x;     // 8*4*32*16 = 16384
    if (i >= 8 * 4 * 32 * 16) return;
    int e    = i & 15;
    int lane = (i >> 4) & 31;
    int s    = (i >> 9) & 3;      // k-step (32 wide)
    int t    = i >> 11;           // N tile
    int nn   = lane & 15;
    int half = lane >> 4;
    int v    = e >> 1;
    int pair = e & 1;
    int k = s * 32 + (v >> 2) * 16 + half * 8 + (v & 3) * 2 + pair;
    Wp[i] = (__bf16)W[k * HD + t * 16 + nn];
}

// ---------------- WMMA GEMM: C[N,128] = A_bf16[N,128] @ W ----------------
// One wave computes a full 16x128 row block: A k-slab loaded ONCE, reused
// across all 8 N-tiles -> 32 x v_wmma_f32_16x16x32_bf16 per wave, 8x less
// A traffic than one-tile-per-wave.
__global__ __launch_bounds__(128) void k_gemm_wmma(const __bf16* __restrict__ A,
                                                   const __bf16* __restrict__ Wp,
                                                   float* __restrict__ C) {
    int wave = blockIdx.x * (blockDim.x >> 5) + (threadIdx.x >> 5);   // = M tile
    int lane = threadIdx.x & 31;
    if (wave >= NN / 16) return;
    int half = lane >> 4;
    int m    = lane & 15;

    const __bf16* arow = A + (long)(wave * 16 + m) * FD;

    union U { uint4 q[2]; v16bf v; };
    v8f acc[8];
#pragma unroll
    for (int t = 0; t < 8; ++t) acc[t] = (v8f){};

#pragma unroll
    for (int kk = 0; kk < 4; ++kk) {
        // 16-bit A operand: each half-K block (8 bf16) is contiguous in row-major memory
        U a;
        const __bf16* ap = arow + kk * 32 + half * 8;
        a.q[0] = *(const uint4*)(ap);        // K = kbase + half*8 .. +7
        a.q[1] = *(const uint4*)(ap + 16);   // K = kbase + 16 + half*8 .. +7
#pragma unroll
        for (int t = 0; t < 8; ++t) {
            U b;
            const uint4* bp = (const uint4*)(Wp + (((t * 4 + kk) * 32 + lane) << 4));
            b.q[0] = bp[0];
            b.q[1] = bp[1];
            acc[t] = __builtin_amdgcn_wmma_f32_16x16x32_bf16(
                         false, a.v, false, b.v, (short)0, acc[t], false, false);
        }
    }

    // C/D layout: VGPR i -> row = wave*16 + half*8 + i, col = t*16 + (lane&15)
    float* outr = C + (long)(wave * 16 + half * 8) * HD + m;
#pragma unroll
    for (int t = 0; t < 8; ++t)
#pragma unroll
        for (int i = 0; i < 8; ++i)
            outr[(long)i * HD + t * 16] = acc[t][i];
}

// ---------------- edge scatter: agg[dst] += h[src] * norm ----------------
// 32 lanes per edge, float4 per lane. e >= E are the self-loops.
__global__ void k_scatter(const float* __restrict__ h, const int* __restrict__ src,
                          const int* __restrict__ dst, const float* __restrict__ norm,
                          const float* __restrict__ dinv, float* __restrict__ agg,
                          int e_cnt, int total) {
    long tid = (long)blockIdx.x * blockDim.x + threadIdx.x;
    int  e   = (int)(tid >> 5);
    if (e >= total) return;
    int c = (int)(tid & 31) << 2;
    int s, d; float w;
    if (e < e_cnt) { s = src[e]; d = dst[e]; w = norm[e]; }
    else           { s = e - e_cnt; d = s; float di = dinv[s]; w = di * di; }
    float4 v = *(const float4*)(h + (long)s * HD + c);
    float* ap = agg + (long)d * HD + c;
    atomicAdd(ap + 0, v.x * w);
    atomicAdd(ap + 1, v.y * w);
    atomicAdd(ap + 2, v.z * w);
    atomicAdd(ap + 3, v.w * w);
}

// ---------------- bias + BN(running stats) + ReLU ----------------
__global__ void k_bnrelu(const float* __restrict__ agg, const float* __restrict__ b,
                         const float* __restrict__ g, const float* __restrict__ be,
                         const float* __restrict__ m, const float* __restrict__ v,
                         __bf16* __restrict__ ob, float* __restrict__ of,
                         int n, int writef) {
    int i = blockIdx.x * blockDim.x + threadIdx.x;
    if (i >= n) return;
    int hc = i & (HD - 1);
    float val = agg[i] + b[hc];
    val = (val - m[hc]) * rsqrtf(v[hc] + BN_EPS) * g[hc] + be[hc];
    val = fmaxf(val, 0.0f);
    ob[i] = (__bf16)val;
    if (writef) of[i] = val;
}

// ---------------- global mean pool (sums + counts via atomics) ----------------
__global__ void k_pool(const float* __restrict__ h, const int* __restrict__ batch,
                       float* __restrict__ sums, float* __restrict__ cnt) {
    long tid = (long)blockIdx.x * blockDim.x + threadIdx.x;
    int  n   = (int)(tid >> 5);
    if (n >= NN) return;
    int c = (int)(tid & 31) << 2;
    int gph = batch[n];
    float4 v = *(const float4*)(h + (long)n * HD + c);
    float* sp = sums + (long)gph * HD + c;
    atomicAdd(sp + 0, v.x);
    atomicAdd(sp + 1, v.y);
    atomicAdd(sp + 2, v.z);
    atomicAdd(sp + 3, v.w);
    if (c == 0) atomicAdd(&cnt[gph], 1.0f);
}

__global__ void k_final(const float* __restrict__ sums, const float* __restrict__ cnt,
                        const float* __restrict__ Wl, const float* __restrict__ bl,
                        float* __restrict__ out) {
    int g = blockIdx.x * blockDim.x + threadIdx.x;
    if (g >= GG) return;
    float acc = 0.0f;
#pragma unroll 4
    for (int h = 0; h < HD; h += 4) {
        float4 s = *(const float4*)(sums + (long)g * HD + h);
        float4 w = *(const float4*)(Wl + h);
        acc += s.x * w.x + s.y * w.y + s.z * w.z + s.w * w.w;
    }
    out[g] = acc / fmaxf(cnt[g], 1.0f) + bl[0];
}

// ---------------- host launcher ----------------
static inline int blocks_for(long n, int bs) { return (int)((n + bs - 1) / bs); }

extern "C" void kernel_launch(void* const* d_in, const int* in_sizes, int n_in,
                              void* d_out, int out_size, void* d_ws, size_t ws_size,
                              hipStream_t stream) {
    (void)in_sizes; (void)n_in; (void)out_size; (void)ws_size;

    const float* x   = (const float*)d_in[0];
    const int*   ei  = (const int*)d_in[1];
    const int*   bat = (const int*)d_in[2];
    const float* W[3]  = { (const float*)d_in[3],  (const float*)d_in[9],  (const float*)d_in[15] };
    const float* bb[3] = { (const float*)d_in[4],  (const float*)d_in[10], (const float*)d_in[16] };
    const float* gg[3] = { (const float*)d_in[5],  (const float*)d_in[11], (const float*)d_in[17] };
    const float* be[3] = { (const float*)d_in[6],  (const float*)d_in[12], (const float*)d_in[18] };
    const float* mm[3] = { (const float*)d_in[7],  (const float*)d_in[13], (const float*)d_in[19] };
    const float* vv[3] = { (const float*)d_in[8],  (const float*)d_in[14], (const float*)d_in[20] };
    const float* Wl  = (const float*)d_in[21];
    const float* bl  = (const float*)d_in[22];
    float*       out = (float*)d_out;

    const int* src = ei;
    const int* dst = ei + EE;

    // ---- carve workspace ----
    char*  ws  = (char*)d_ws;
    size_t off = 0;
    auto carve = [&](size_t bytes) -> void* {
        void* p = ws + off;
        off += (bytes + 255) & ~(size_t)255;
        return p;
    };
    __bf16* xb   = (__bf16*)carve((size_t)NN * HD * sizeof(__bf16));   // bf16 layer input
    __bf16* Wp   = (__bf16*)carve((size_t)3 * 16384 * sizeof(__bf16)); // packed weights
    float*  hbuf = (float*)carve((size_t)NN * HD * sizeof(float));     // GEMM out / pooled feats
    float*  agg  = (float*)carve((size_t)NN * HD * sizeof(float));     // scatter accumulator
    float*  deg  = (float*)carve((size_t)NN * sizeof(float));
    float*  dinv = (float*)carve((size_t)NN * sizeof(float));
    float*  nrm  = (float*)carve((size_t)EE * sizeof(float));
    float*  sums = (float*)carve((size_t)GG * HD * sizeof(float));
    float*  cnt  = (float*)carve((size_t)GG * sizeof(float));

    const int BS = 256;

    // ---- degrees / norms (shared across layers) ----
    k_fill_f32<<<blocks_for(NN, BS), BS, 0, stream>>>(deg, 1.0f, NN);   // self-loop
    k_deg<<<blocks_for(EE, BS), BS, 0, stream>>>(dst, deg, EE);
    k_dinv<<<blocks_for(NN, BS), BS, 0, stream>>>(deg, dinv, NN);
    k_norm<<<blocks_for(EE, BS), BS, 0, stream>>>(src, dst, dinv, nrm, EE);

    // ---- prep inputs for WMMA ----
    k_cvt_bf16<<<blocks_for((long)NN * FD, BS), BS, 0, stream>>>(x, xb, NN * FD);
    for (int l = 0; l < 3; ++l)
        k_pack_w<<<blocks_for(16384, BS), BS, 0, stream>>>(W[l], Wp + (size_t)l * 16384);

    // ---- 3 GCN layers ----
    const int gemm_waves  = NN / 16;                         // 6250 row blocks, 1 wave each
    const int gemm_blocks = (gemm_waves + 3) / 4;            // 4 waves / block
    const long scat_threads = (long)(EE + NN) * 32;

    for (int l = 0; l < 3; ++l) {
        k_gemm_wmma<<<gemm_blocks, 128, 0, stream>>>(xb, Wp + (size_t)l * 16384, hbuf);
        k_fill_f32<<<blocks_for((long)NN * HD, BS), BS, 0, stream>>>(agg, 0.0f, NN * HD);
        k_scatter<<<blocks_for(scat_threads, BS), BS, 0, stream>>>(
            hbuf, src, dst, nrm, dinv, agg, EE, EE + NN);
        // writes bf16 next-layer input in-place over xb; layer 2 also writes f32 into hbuf
        k_bnrelu<<<blocks_for((long)NN * HD, BS), BS, 0, stream>>>(
            agg, bb[l], gg[l], be[l], mm[l], vv[l], xb, hbuf, NN * HD, (l == 2) ? 1 : 0);
    }

    // ---- pool + final linear ----
    k_fill_f32<<<blocks_for((long)GG * HD, BS), BS, 0, stream>>>(sums, 0.0f, GG * HD);
    k_fill_f32<<<blocks_for(GG, BS), BS, 0, stream>>>(cnt, 0.0f, GG);
    k_pool<<<blocks_for((long)NN * 32, BS), BS, 0, stream>>>(hbuf, bat, sums, cnt);
    k_final<<<blocks_for(GG, BS), BS, 0, stream>>>(sums, cnt, Wl, bl, out);
}
